// FAVORiserBlock_56616258895966
// MI455X (gfx1250) — matrix operand order repaired and checked by reading, not compiled, not run on hardware
//
#include <hip/hip_runtime.h>
#include <math.h>

// ---------------- problem dims ----------------
#define BATCHN   4
#define SEQN     2048
#define TTOK     (BATCHN*SEQN)      // 8192 tokens
#define DMODEL   1024
#define DFFN     4096
#define HEADS_N  8
#define DHEAD    64
#define INNERD   (HEADS_N*DHEAD)    // 512
#define PFFD     4096
#define MFEAT    266                // FAVOR+ features
#define MPAD     288                // padded to mult of 32 for K dim
#define ZHB      (HEADS_N*BATCHN)   // 32 (head,batch) slabs

#define DN_CONST    0.3535533905932738f   // 64^-0.25
#define RATIO_CONST 0.0613139339484966f   // 266^-0.5
#define EPSK_CONST  1e-4f
#define EPSLN_CONST 1e-5f

typedef __attribute__((ext_vector_type(8)))  float  v8f;
typedef __attribute__((ext_vector_type(16))) __bf16 v16bf;
typedef int v4i __attribute__((vector_size(16)));   // matches async-LDS builtin param type

union FragAB { v16bf v; uint4 q[2]; };

#define AS1 __attribute__((address_space(1)))
#define AS3 __attribute__((address_space(3)))

// async global->LDS copy path (CDNA5 data mover, ASYNCcnt), with safe fallback
#if defined(__gfx1250__) && __has_builtin(__builtin_amdgcn_global_load_async_to_lds_b128) && \
    __has_builtin(__builtin_amdgcn_s_wait_asynccnt)
#define USE_ASYNC_COPY 1
#else
#define USE_ASYNC_COPY 0
#endif

static __device__ __forceinline__ unsigned short f2bf(float f) {
  unsigned int u = __float_as_uint(f);
  u += 0x7FFFu + ((u >> 16) & 1u);          // round-to-nearest-even
  return (unsigned short)(u >> 16);
}
static __device__ __forceinline__ float bf2f(unsigned short h) {
  return __uint_as_float(((unsigned int)h) << 16);
}
static __device__ __forceinline__ float gelu_exact(float x) {
  return 0.5f * x * (1.0f + erff(x * 0.7071067811865475f));
}

// ---------------- WMMA GEMM:  C[M,N] = A[M,K](bf16) * Bt[N,K](bf16)^T  ----------------
// Block tile 128x128, K-step 32, 8 waves (wave32): wave grid 2(M)x4(N), each wave 64x32.
#define TM 128
#define TN 128
#define TKS 32
#define APITCH 48   // bf16 elems per LDS row (pad; 96B rows, 16B aligned chunks)

enum { EPI_NONE = 0, EPI_BIAS = 1, EPI_BIAS_GELU = 2, EPI_BIAS_RES = 3, EPI_ROWSCALE = 4 };

template<int EPI>
__global__ __launch_bounds__(256)
void gemm_bf16_wmma(const unsigned short* __restrict__ A, int lda, unsigned long long strideA,
                    const unsigned short* __restrict__ Bt, int ldb, unsigned long long strideB,
                    float* __restrict__ outF, unsigned short* __restrict__ outB,
                    unsigned long long strideC,
                    const float* __restrict__ bias,
                    const float* __restrict__ resid, unsigned long long strideR,
                    const float* __restrict__ rowscale, unsigned long long strideS,
                    int M, int N, int K)
{
#if USE_ASYNC_COPY
  __shared__ __align__(16) unsigned short As[2][TM * APITCH];
  __shared__ __align__(16) unsigned short Bs[2][TN * APITCH];
#else
  __shared__ __align__(16) unsigned short As[1][TM * APITCH];
  __shared__ __align__(16) unsigned short Bs[1][TN * APITCH];
#endif

  A  += (size_t)blockIdx.z * strideA;
  Bt += (size_t)blockIdx.z * strideB;
  const size_t cOff = (size_t)blockIdx.z * strideC;

  const int tid   = threadIdx.x;
  const int wave  = tid >> 5;
  const int lane  = tid & 31;
  const int l15   = lane & 15;
  const int khalf = lane >> 4;
  const int waveM = (wave & 1) * 64;
  const int waveN = (wave >> 1) * 32;
  const int tileM = blockIdx.y * TM;
  const int tileN = blockIdx.x * TN;

  // staging role (loop-invariant): each thread owns 32B of one A row and one B row
  const int ldRow  = tid >> 1;   // 0..127
  const int ldHalf = tid & 1;    // 0..1 (16 bf16 each)
  const bool aValid = (tileM + ldRow) < M;
  const bool bValid = (tileN + ldRow) < N;
  const unsigned short* Arow = A + (size_t)(tileM + ldRow) * lda + ldHalf * 16;
  const unsigned short* Brow = Bt + (size_t)(tileN + ldRow) * ldb + ldHalf * 16;
  const int ldsOff = ldRow * APITCH + ldHalf * 16;

  v8f acc[4][2];
  const v8f vzero = {0.f, 0.f, 0.f, 0.f, 0.f, 0.f, 0.f, 0.f};
#pragma unroll
  for (int i = 0; i < 4; i++)
#pragma unroll
    for (int j = 0; j < 2; j++) acc[i][j] = vzero;

#if USE_ASYNC_COPY
  // zero-fill OOB LDS rows once (pattern is K-invariant), both buffers
  {
    const uint4 z = {0, 0, 0, 0};
    if (!aValid) {
      *reinterpret_cast<uint4*>(&As[0][ldsOff]) = z;
      *reinterpret_cast<uint4*>(&As[0][ldsOff + 8]) = z;
      *reinterpret_cast<uint4*>(&As[1][ldsOff]) = z;
      *reinterpret_cast<uint4*>(&As[1][ldsOff + 8]) = z;
    }
    if (!bValid) {
      *reinterpret_cast<uint4*>(&Bs[0][ldsOff]) = z;
      *reinterpret_cast<uint4*>(&Bs[0][ldsOff + 8]) = z;
      *reinterpret_cast<uint4*>(&Bs[1][ldsOff]) = z;
      *reinterpret_cast<uint4*>(&Bs[1][ldsOff + 8]) = z;
    }
  }
  auto issue_stage = [&](int buf, int k0) {
    if (aValid) {
      __builtin_amdgcn_global_load_async_to_lds_b128(
          (AS1 v4i*)(Arow + k0), (AS3 v4i*)&As[buf][ldsOff], 0, 0);
      __builtin_amdgcn_global_load_async_to_lds_b128(
          (AS1 v4i*)(Arow + k0 + 8), (AS3 v4i*)&As[buf][ldsOff + 8], 0, 0);
    }
    if (bValid) {
      __builtin_amdgcn_global_load_async_to_lds_b128(
          (AS1 v4i*)(Brow + k0), (AS3 v4i*)&Bs[buf][ldsOff], 0, 0);
      __builtin_amdgcn_global_load_async_to_lds_b128(
          (AS1 v4i*)(Brow + k0 + 8), (AS3 v4i*)&Bs[buf][ldsOff + 8], 0, 0);
    }
  };
  issue_stage(0, 0);
  int buf = 0;
#endif

  for (int k0 = 0; k0 < K; k0 += TKS) {
#if USE_ASYNC_COPY
    const bool hasNext = (k0 + TKS) < K;
    if (hasNext) issue_stage(buf ^ 1, k0 + TKS);   // stream next tile while computing
    // wait only for THIS buffer's async ops (next stage may stay in flight)
    if (hasNext) {
      if (aValid && bValid)      __builtin_amdgcn_s_wait_asynccnt(4);
      else if (aValid || bValid) __builtin_amdgcn_s_wait_asynccnt(2);
      else                       __builtin_amdgcn_s_wait_asynccnt(0);
    } else {
      __builtin_amdgcn_s_wait_asynccnt(0);
    }
    __syncthreads();
    const unsigned short* Ab = &As[buf][0];
    const unsigned short* Bb = &Bs[buf][0];
#else
    // ---- synchronous staging fallback ----
    {
      uint4 a0 = {0, 0, 0, 0}, a1 = {0, 0, 0, 0};
      if (aValid) {
        const uint4* gp = reinterpret_cast<const uint4*>(Arow + k0);
        a0 = gp[0]; a1 = gp[1];
        if (k0 + TKS < K) __builtin_prefetch(Arow + k0 + TKS, 0, 1);
      }
      uint4* sp = reinterpret_cast<uint4*>(&As[0][ldsOff]);
      sp[0] = a0; sp[1] = a1;
      uint4 b0 = {0, 0, 0, 0}, b1 = {0, 0, 0, 0};
      if (bValid) {
        const uint4* gp = reinterpret_cast<const uint4*>(Brow + k0);
        b0 = gp[0]; b1 = gp[1];
        if (k0 + TKS < K) __builtin_prefetch(Brow + k0 + TKS, 0, 1);
      }
      uint4* spb = reinterpret_cast<uint4*>(&Bs[0][ldsOff]);
      spb[0] = b0; spb[1] = b1;
    }
    __syncthreads();
    const unsigned short* Ab = &As[0][0];
    const unsigned short* Bb = &Bs[0][0];
#endif

    // ---- B fragments: lane holds col = base + (lane&15), K = khalf*16 .. +15 (contig) ----
    FragAB bfr[2];
#pragma unroll
    for (int j = 0; j < 2; j++) {
      const int col = waveN + j * 16 + l15;
      const uint4* sp = reinterpret_cast<const uint4*>(&Bb[col * APITCH + khalf * 16]);
      bfr[j].q[0] = sp[0]; bfr[j].q[1] = sp[1];
    }
    // ---- A fragments: lane holds row, K = khalf*16 .. +15 (contig) ----
#pragma unroll
    for (int i = 0; i < 4; i++) {
      const int row = waveM + i * 16 + l15;
      FragAB af;
      af.q[0] = *reinterpret_cast<const uint4*>(&Ab[row * APITCH + khalf * 16]);
      af.q[1] = *reinterpret_cast<const uint4*>(&Ab[row * APITCH + khalf * 16 + 8]);
#pragma unroll
      for (int j = 0; j < 2; j++) {
        acc[i][j] = __builtin_amdgcn_wmma_f32_16x16x32_bf16(
            false, af.v, false, bfr[j].v, (short)0, acc[i][j], false, false);
      }
    }
    __syncthreads();
#if USE_ASYNC_COPY
    buf ^= 1;
#endif
  }

  // ---- epilogue + store. C layout: VGPR r -> M=r (lanes 0-15), M=r+8 (lanes 16-31) ----
#pragma unroll
  for (int i = 0; i < 4; i++) {
#pragma unroll
    for (int j = 0; j < 2; j++) {
      const int col = tileN + waveN + j * 16 + l15;
      if (col >= N) continue;
#pragma unroll
      for (int r = 0; r < 8; r++) {
        const int row = tileM + waveM + i * 16 + r + khalf * 8;
        if (row >= M) continue;
        float val = acc[i][j][r];
        if (EPI == EPI_BIAS)       val += bias[col];
        else if (EPI == EPI_BIAS_GELU) { val += bias[col]; val = gelu_exact(val); }
        else if (EPI == EPI_BIAS_RES)  { val += bias[col] +
                 resid[(size_t)blockIdx.z * strideR + (size_t)row * N + col]; }
        else if (EPI == EPI_ROWSCALE)  { val *= rowscale[(size_t)blockIdx.z * strideS + row]; }
        const size_t idx = cOff + (size_t)row * N + col;
        if (outF) outF[idx] = val;
        if (outB) outB[idx] = f2bf(val);
      }
    }
  }
}

// ---------------- LayerNorm: one block (256 thr) per token row of 1024 ----------------
__global__ __launch_bounds__(256)
void ln_kernel(const float* __restrict__ x, const float* __restrict__ g,
               const float* __restrict__ b, float* __restrict__ outF,
               unsigned short* __restrict__ outB)
{
  __shared__ float red[256];
  const int t = blockIdx.x;
  const float* row = x + (size_t)t * DMODEL;
  float s = 0.f;
  for (int i = threadIdx.x; i < DMODEL; i += 256) s += row[i];
  red[threadIdx.x] = s; __syncthreads();
  for (int o = 128; o > 0; o >>= 1) { if (threadIdx.x < o) red[threadIdx.x] += red[threadIdx.x + o]; __syncthreads(); }
  const float mu = red[0] * (1.0f / DMODEL);
  __syncthreads();
  float v = 0.f;
  for (int i = threadIdx.x; i < DMODEL; i += 256) { float d = row[i] - mu; v += d * d; }
  red[threadIdx.x] = v; __syncthreads();
  for (int o = 128; o > 0; o >>= 1) { if (threadIdx.x < o) red[threadIdx.x] += red[threadIdx.x + o]; __syncthreads(); }
  const float rs = rsqrtf(red[0] * (1.0f / DMODEL) + EPSLN_CONST);
  for (int i = threadIdx.x; i < DMODEL; i += 256) {
    const float y = (row[i] - mu) * rs * g[i] + b[i];
    const size_t idx = (size_t)t * DMODEL + i;
    if (outF) outF[idx] = y;
    if (outB) outB[idx] = f2bf(y);
  }
}

// ---- weight transpose+convert: W[K][N] (f32) -> Wt[N][K] (bf16) ----
__global__ void wt_kernel(const float* __restrict__ W, unsigned short* __restrict__ Wt,
                          int K, int N)
{
  const size_t total = (size_t)K * N;
  for (size_t i = (size_t)blockIdx.x * blockDim.x + threadIdx.x; i < total;
       i += (size_t)gridDim.x * blockDim.x) {
    const size_t k = i / N, n = i % N;
    Wt[n * (size_t)K + k] = f2bf(W[i]);
  }
}

__global__ void cvt_kernel(const float* __restrict__ src, unsigned short* __restrict__ dst,
                           size_t n)
{
  for (size_t i = (size_t)blockIdx.x * blockDim.x + threadIdx.x; i < n;
       i += (size_t)gridDim.x * blockDim.x)
    dst[i] = f2bf(src[i]);
}

// ---- split heads: q,k scaled by d^-1/4 -> qh/kh[h][T][64]; vvT[z][64][S]; diag = 0.5*||q*dn||^2 ----
__global__ __launch_bounds__(64)
void split_heads_kernel(const float* __restrict__ q, const float* __restrict__ k,
                        const float* __restrict__ v,
                        unsigned short* __restrict__ qh, unsigned short* __restrict__ kh,
                        unsigned short* __restrict__ vvT,
                        float* __restrict__ diag_q, float* __restrict__ diag_k)
{
  __shared__ float red[64];
  const int t = blockIdx.x, h = blockIdx.y, d = threadIdx.x;
  const size_t src = (size_t)t * INNERD + h * DHEAD + d;
  const float qv = q[src] * DN_CONST;
  const float kv = k[src] * DN_CONST;
  const float vv = v[src];
  const size_t ht = (size_t)h * TTOK + t;
  qh[ht * DHEAD + d] = f2bf(qv);
  kh[ht * DHEAD + d] = f2bf(kv);
  const int b = t / SEQN, n = t % SEQN;
  const size_t z = (size_t)h * BATCHN + b;
  vvT[(z * DHEAD + d) * SEQN + n] = f2bf(vv);

  red[d] = qv * qv; __syncthreads();
  for (int o = 32; o > 0; o >>= 1) { if (d < o) red[d] += red[d + o]; __syncthreads(); }
  if (d == 0) diag_q[ht] = 0.5f * red[0];
  __syncthreads();
  red[d] = kv * kv; __syncthreads();
  for (int o = 32; o > 0; o >>= 1) { if (d < o) red[d] += red[d + o]; __syncthreads(); }
  if (d == 0) diag_k[ht] = 0.5f * red[0];
}

// ---- query features: qp[h][t][0..287] = ratio*(exp(dd - diag - rowmax)+eps), pad zeros ----
__global__ __launch_bounds__(256)
void qp_kernel(const float* __restrict__ dd, const float* __restrict__ diag,
               unsigned short* __restrict__ qp)
{
  __shared__ float red[256];
  const int t = blockIdx.x, h = blockIdx.y;
  const size_t base = ((size_t)h * TTOK + t) * MFEAT;
  float m = -1e30f;
  for (int i = threadIdx.x; i < MFEAT; i += 256) m = fmaxf(m, dd[base + i]);
  red[threadIdx.x] = m; __syncthreads();
  for (int o = 128; o > 0; o >>= 1) { if (threadIdx.x < o) red[threadIdx.x] = fmaxf(red[threadIdx.x], red[threadIdx.x + o]); __syncthreads(); }
  const float sub = red[0] + diag[(size_t)h * TTOK + t];
  const size_t ob = ((size_t)h * TTOK + t) * MPAD;
  for (int i = threadIdx.x; i < MPAD; i += 256) {
    float val = 0.f;
    if (i < MFEAT) val = RATIO_CONST * (expf(dd[base + i] - sub) + EPSK_CONST);
    qp[ob + i] = f2bf(val);
  }
}

// ---- per-(h,b) slab max of dd over [S x MFEAT] ----
__global__ __launch_bounds__(256)
void kmax_kernel(const float* __restrict__ dd, float* __restrict__ kmax)
{
  __shared__ float red[256];
  const int z = blockIdx.x;
  const size_t base = (size_t)z * SEQN * MFEAT;
  float m = -1e30f;
  for (size_t i = threadIdx.x; i < (size_t)SEQN * MFEAT; i += 256) m = fmaxf(m, dd[base + i]);
  red[threadIdx.x] = m; __syncthreads();
  for (int o = 128; o > 0; o >>= 1) { if (threadIdx.x < o) red[threadIdx.x] = fmaxf(red[threadIdx.x], red[threadIdx.x + o]); __syncthreads(); }
  if (threadIdx.x == 0) kmax[z] = red[0];
}

// ---- key features, written transposed: kpT[z][m][n] ----
__global__ void kpt_kernel(const float* __restrict__ dd, const float* __restrict__ diag,
                           const float* __restrict__ kmax, unsigned short* __restrict__ kpT)
{
  const size_t total = (size_t)ZHB * MFEAT * SEQN;
  for (size_t i = (size_t)blockIdx.x * blockDim.x + threadIdx.x; i < total;
       i += (size_t)gridDim.x * blockDim.x) {
    const size_t z = i / ((size_t)MFEAT * SEQN);
    const size_t r = i % ((size_t)MFEAT * SEQN);
    const int m = (int)(r / SEQN), n = (int)(r % SEQN);
    const float v = RATIO_CONST *
        (expf(dd[z * SEQN * MFEAT + (size_t)n * MFEAT + m] - diag[z * SEQN + n] - kmax[z]) +
         EPSK_CONST);
    kpT[i] = f2bf(v);
  }
}

// ---- k_sum[z][m] = sum_n kpT[z][m][n] ----
__global__ __launch_bounds__(256)
void ksum_kernel(const unsigned short* __restrict__ kpT, float* __restrict__ ksum)
{
  __shared__ float red[256];
  const int m = blockIdx.x, z = blockIdx.y;
  const size_t base = ((size_t)z * MFEAT + m) * SEQN;
  float s = 0.f;
  for (int n = threadIdx.x; n < SEQN; n += 256) s += bf2f(kpT[base + n]);
  red[threadIdx.x] = s; __syncthreads();
  for (int o = 128; o > 0; o >>= 1) { if (threadIdx.x < o) red[threadIdx.x] += red[threadIdx.x + o]; __syncthreads(); }
  if (threadIdx.x == 0) ksum[(size_t)z * MFEAT + m] = red[0];
}

// ---- d_inv[h][t] = 1 / (qp[t,:] . k_sum[z,:]) ----
__global__ __launch_bounds__(256)
void dinv_kernel(const unsigned short* __restrict__ qp, const float* __restrict__ ksum,
                 float* __restrict__ dinv)
{
  __shared__ float red[256];
  const int t = blockIdx.x, h = blockIdx.y;
  const int z = h * BATCHN + t / SEQN;
  const size_t qb = ((size_t)h * TTOK + t) * MPAD;
  float s = 0.f;
  for (int m = threadIdx.x; m < MFEAT; m += 256) s += bf2f(qp[qb + m]) * ksum[(size_t)z * MFEAT + m];
  red[threadIdx.x] = s; __syncthreads();
  for (int o = 128; o > 0; o >>= 1) { if (threadIdx.x < o) red[threadIdx.x] += red[threadIdx.x + o]; __syncthreads(); }
  if (threadIdx.x == 0) dinv[(size_t)h * TTOK + t] = 1.0f / red[0];
}

// ---- ctx[z][266][64] (f32) -> ctxT[z][64][288] (bf16, K-padded) ----
__global__ void ctxt_kernel(const float* __restrict__ ctx, unsigned short* __restrict__ ctxT)
{
  const size_t total = (size_t)ZHB * DHEAD * MPAD;
  for (size_t i = (size_t)blockIdx.x * blockDim.x + threadIdx.x; i < total;
       i += (size_t)gridDim.x * blockDim.x) {
    const size_t z = i / ((size_t)DHEAD * MPAD);
    const size_t r = i % ((size_t)DHEAD * MPAD);
    const int d = (int)(r / MPAD), m = (int)(r % MPAD);
    const float v = (m < MFEAT) ? ctx[(z * MFEAT + m) * DHEAD + d] : 0.f;
    ctxT[i] = f2bf(v);
  }
}

// ---- merge heads: o[z][n][64] -> obf[t][512] (bf16) ----
__global__ void merge_kernel(const float* __restrict__ o, unsigned short* __restrict__ obf)
{
  const size_t total = (size_t)TTOK * INNERD;
  for (size_t i = (size_t)blockIdx.x * blockDim.x + threadIdx.x; i < total;
       i += (size_t)gridDim.x * blockDim.x) {
    const int t = (int)(i / INNERD), c = (int)(i % INNERD);
    const int h = c / DHEAD, d = c % DHEAD;
    const int b = t / SEQN, n = t % SEQN;
    const size_t z = (size_t)h * BATCHN + b;
    obf[i] = f2bf(o[(z * SEQN + n) * DHEAD + d]);
  }
}

// ---- gating: xg = u * v + x ----
__global__ void gate_kernel(const float* __restrict__ u, const float* __restrict__ v,
                            const float* __restrict__ x, float* __restrict__ xg)
{
  const size_t total = (size_t)TTOK * DMODEL;
  for (size_t i = (size_t)blockIdx.x * blockDim.x + threadIdx.x; i < total;
       i += (size_t)gridDim.x * blockDim.x)
    xg[i] = u[i] * v[i] + x[i];
}

// ---------------- host orchestration ----------------
static inline void launch_gemm(hipStream_t s, int epi,
    const unsigned short* A, int lda, unsigned long long sA,
    const unsigned short* Bt, int ldb, unsigned long long sB,
    float* outF, unsigned short* outB, unsigned long long sC,
    const float* bias, const float* resid, unsigned long long sR,
    const float* rows, unsigned long long sS, int M, int N, int K, int nb)
{
  dim3 grid((N + TN - 1) / TN, (M + TM - 1) / TM, nb);
  dim3 blk(256);
  switch (epi) {
    case EPI_NONE:
      gemm_bf16_wmma<EPI_NONE><<<grid, blk, 0, s>>>(A, lda, sA, Bt, ldb, sB, outF, outB, sC,
                                                    bias, resid, sR, rows, sS, M, N, K); break;
    case EPI_BIAS:
      gemm_bf16_wmma<EPI_BIAS><<<grid, blk, 0, s>>>(A, lda, sA, Bt, ldb, sB, outF, outB, sC,
                                                    bias, resid, sR, rows, sS, M, N, K); break;
    case EPI_BIAS_GELU:
      gemm_bf16_wmma<EPI_BIAS_GELU><<<grid, blk, 0, s>>>(A, lda, sA, Bt, ldb, sB, outF, outB, sC,
                                                         bias, resid, sR, rows, sS, M, N, K); break;
    case EPI_BIAS_RES:
      gemm_bf16_wmma<EPI_BIAS_RES><<<grid, blk, 0, s>>>(A, lda, sA, Bt, ldb, sB, outF, outB, sC,
                                                        bias, resid, sR, rows, sS, M, N, K); break;
    default:
      gemm_bf16_wmma<EPI_ROWSCALE><<<grid, blk, 0, s>>>(A, lda, sA, Bt, ldb, sB, outF, outB, sC,
                                                        bias, resid, sR, rows, sS, M, N, K); break;
  }
}

extern "C" void kernel_launch(void* const* d_in, const int* in_sizes, int n_in,
                              void* d_out, int out_size, void* d_ws, size_t ws_size,
                              hipStream_t stream)
{
  const float* x      = (const float*)d_in[0];
  const float* ln_g   = (const float*)d_in[1];
  const float* ln_b   = (const float*)d_in[2];
  const float* proj_W = (const float*)d_in[3];
  const float* proj_b = (const float*)d_in[4];
  const float* a_ln_g = (const float*)d_in[5];
  const float* a_ln_b = (const float*)d_in[6];
  const float* Wq     = (const float*)d_in[7];
  const float* bq     = (const float*)d_in[8];
  const float* Wk     = (const float*)d_in[9];
  const float* bk     = (const float*)d_in[10];
  const float* Wv     = (const float*)d_in[11];
  const float* bv     = (const float*)d_in[12];
  const float* Wo     = (const float*)d_in[13];
  const float* bo     = (const float*)d_in[14];
  const float* projM  = (const float*)d_in[15];
  const float* f_ln_g = (const float*)d_in[16];
  const float* f_ln_b = (const float*)d_in[17];
  const float* pW1    = (const float*)d_in[18];
  const float* pb1    = (const float*)d_in[19];
  const float* pW2    = (const float*)d_in[20];
  const float* pb2    = (const float*)d_in[21];
  const float* Wf1    = (const float*)d_in[22];
  const float* bf1    = (const float*)d_in[23];
  const float* Wf2    = (const float*)d_in[24];
  const float* bf2    = (const float*)d_in[25];

  // --- workspace bump allocator (256B aligned) ---
  char* base = (char*)d_ws;
  size_t off = 0;
  auto alloc = [&](size_t bytes) -> void* {
    off = (off + 255) & ~(size_t)255;
    void* p = base + off;
    off += bytes;
    return p;
  };
  const size_t TD = (size_t)TTOK * DMODEL, TI = (size_t)TTOK * INNERD,
               TF = (size_t)TTOK * DFFN, HT = (size_t)HEADS_N * TTOK;

  // fp32 buffers
  float* y    = (float*)alloc(TD * 4);
  float* u    = (float*)alloc(TD * 4);
  float* v    = (float*)alloc(TD * 4);
  float* xg   = (float*)alloc(TD * 4);
  float* qf   = (float*)alloc(TI * 4);
  float* kf   = (float*)alloc(TI * 4);
  float* vf   = (float*)alloc(TI * 4);
  float* dd   = (float*)alloc(HT * MFEAT * 4);
  float* ctx  = (float*)alloc((size_t)ZHB * MFEAT * DHEAD * 4);
  float* obuf = (float*)alloc(HT * DHEAD * 4);
  float* diag_q = (float*)alloc(HT * 4);
  float* diag_k = (float*)alloc(HT * 4);
  float* kmaxb  = (float*)alloc(ZHB * 4);
  float* ksumb  = (float*)alloc((size_t)ZHB * MFEAT * 4);
  float* dinvb  = (float*)alloc(HT * 4);
  // bf16 buffers
  unsigned short* ybf   = (unsigned short*)alloc(TD * 2);
  unsigned short* yabf  = (unsigned short*)alloc(TD * 2);
  unsigned short* yfbf  = (unsigned short*)alloc(TD * 2);
  unsigned short* qh    = (unsigned short*)alloc(HT * DHEAD * 2);
  unsigned short* kh    = (unsigned short*)alloc(HT * DHEAD * 2);
  unsigned short* vvT   = (unsigned short*)alloc(HT * DHEAD * 2);
  unsigned short* qpbf  = (unsigned short*)alloc(HT * MPAD * 2);
  unsigned short* kpT   = (unsigned short*)alloc(HT * MFEAT * 2);
  unsigned short* ctxT  = (unsigned short*)alloc((size_t)ZHB * DHEAD * MPAD * 2);
  unsigned short* obf   = (unsigned short*)alloc(TI * 2);
  unsigned short* h1bf  = (unsigned short*)alloc(TF * 2);
  unsigned short* projbf = (unsigned short*)alloc((size_t)MFEAT * DHEAD * 2);
  unsigned short* projWT = (unsigned short*)alloc((size_t)DMODEL * DMODEL * 2);
  unsigned short* WqT   = (unsigned short*)alloc((size_t)INNERD * DMODEL * 2);
  unsigned short* WkT   = (unsigned short*)alloc((size_t)INNERD * DMODEL * 2);
  unsigned short* WvT   = (unsigned short*)alloc((size_t)INNERD * DMODEL * 2);
  unsigned short* WoT   = (unsigned short*)alloc((size_t)DMODEL * INNERD * 2);
  unsigned short* pW1T  = (unsigned short*)alloc((size_t)PFFD * DMODEL * 2);
  unsigned short* pW2T  = (unsigned short*)alloc((size_t)DMODEL * PFFD * 2);
  unsigned short* Wf1T  = (unsigned short*)alloc((size_t)DFFN * DMODEL * 2);
  unsigned short* Wf2T  = (unsigned short*)alloc((size_t)DMODEL * DFFN * 2);
  (void)ws_size; (void)in_sizes; (void)n_in; (void)out_size;

  // --- per-launch weight prep (deterministic, graph-safe) ---
  wt_kernel<<<2048, 256, 0, stream>>>(proj_W, projWT, DMODEL, DMODEL);
  wt_kernel<<<2048, 256, 0, stream>>>(Wq, WqT, DMODEL, INNERD);
  wt_kernel<<<2048, 256, 0, stream>>>(Wk, WkT, DMODEL, INNERD);
  wt_kernel<<<2048, 256, 0, stream>>>(Wv, WvT, DMODEL, INNERD);
  wt_kernel<<<2048, 256, 0, stream>>>(Wo, WoT, INNERD, DMODEL);
  cvt_kernel<<<128, 256, 0, stream>>>(projM, projbf, (size_t)MFEAT * DHEAD);
  wt_kernel<<<4096, 256, 0, stream>>>(pW1, pW1T, DMODEL, PFFD);
  wt_kernel<<<4096, 256, 0, stream>>>(pW2, pW2T, PFFD, DMODEL);
  wt_kernel<<<4096, 256, 0, stream>>>(Wf1, Wf1T, DMODEL, DFFN);
  wt_kernel<<<4096, 256, 0, stream>>>(Wf2, Wf2T, DFFN, DMODEL);

  // --- 1. y = LN(x); u = y @ proj_W + proj_b ---
  ln_kernel<<<TTOK, 256, 0, stream>>>(x, ln_g, ln_b, y, ybf);
  launch_gemm(stream, EPI_BIAS, ybf, DMODEL, 0, projWT, DMODEL, 0,
              u, nullptr, 0, proj_b, nullptr, 0, nullptr, 0, TTOK, DMODEL, DMODEL, 1);

  // --- 2. Performer attention on y ---
  ln_kernel<<<TTOK, 256, 0, stream>>>(y, a_ln_g, a_ln_b, nullptr, yabf);
  launch_gemm(stream, EPI_BIAS, yabf, DMODEL, 0, WqT, DMODEL, 0,
              qf, nullptr, 0, bq, nullptr, 0, nullptr, 0, TTOK, INNERD, DMODEL, 1);
  launch_gemm(stream, EPI_BIAS, yabf, DMODEL, 0, WkT, DMODEL, 0,
              kf, nullptr, 0, bk, nullptr, 0, nullptr, 0, TTOK, INNERD, DMODEL, 1);
  launch_gemm(stream, EPI_BIAS, yabf, DMODEL, 0, WvT, DMODEL, 0,
              vf, nullptr, 0, bv, nullptr, 0, nullptr, 0, TTOK, INNERD, DMODEL, 1);
  split_heads_kernel<<<dim3(TTOK, HEADS_N), 64, 0, stream>>>(qf, kf, vf, qh, kh, vvT,
                                                             diag_q, diag_k);
  // dd_q = (q*dn) @ proj^T, per head
  launch_gemm(stream, EPI_NONE, qh, DHEAD, (unsigned long long)TTOK * DHEAD,
              projbf, DHEAD, 0, dd, nullptr, (unsigned long long)TTOK * MFEAT,
              nullptr, nullptr, 0, nullptr, 0, TTOK, MFEAT, DHEAD, HEADS_N);
  qp_kernel<<<dim3(TTOK, HEADS_N), 256, 0, stream>>>(dd, diag_q, qpbf);
  // dd_k (reuse dd)
  launch_gemm(stream, EPI_NONE, kh, DHEAD, (unsigned long long)TTOK * DHEAD,
              projbf, DHEAD, 0, dd, nullptr, (unsigned long long)TTOK * MFEAT,
              nullptr, nullptr, 0, nullptr, 0, TTOK, MFEAT, DHEAD, HEADS_N);
  kmax_kernel<<<ZHB, 256, 0, stream>>>(dd, kmaxb);
  kpt_kernel<<<4096, 256, 0, stream>>>(dd, diag_k, kmaxb, kpT);
  ksum_kernel<<<dim3(MFEAT, ZHB), 256, 0, stream>>>(kpT, ksumb);
  // ctx[z] = kp^T @ vv : [266,2048] x [2048,64]
  launch_gemm(stream, EPI_NONE, kpT, SEQN, (unsigned long long)MFEAT * SEQN,
              vvT, SEQN, (unsigned long long)DHEAD * SEQN,
              ctx, nullptr, (unsigned long long)MFEAT * DHEAD,
              nullptr, nullptr, 0, nullptr, 0, MFEAT, DHEAD, SEQN, ZHB);
  ctxt_kernel<<<1024, 256, 0, stream>>>(ctx, ctxT);
  dinv_kernel<<<dim3(TTOK, HEADS_N), 256, 0, stream>>>(qpbf, ksumb, dinvb);
  // o[z] = d_inv * (qp @ ctx) : [2048,288] x [288,64], row-scale epilogue
  launch_gemm(stream, EPI_ROWSCALE, qpbf, MPAD, (unsigned long long)SEQN * MPAD,
              ctxT, MPAD, (unsigned long long)DHEAD * MPAD,
              obuf, nullptr, (unsigned long long)SEQN * DHEAD,
              nullptr, nullptr, 0, dinvb, (unsigned long long)SEQN, SEQN, DHEAD, MPAD, ZHB);
  merge_kernel<<<4096, 256, 0, stream>>>(obuf, obf);
  // v = y + o @ Wo + bo
  launch_gemm(stream, EPI_BIAS_RES, obf, INNERD, 0, WoT, INNERD, 0,
              v, nullptr, 0, bo, y, 0, nullptr, 0, TTOK, DMODEL, INNERD, 1);
  // performer FF: v = v + gelu(LN(v) @ pW1 + pb1) @ pW2 + pb2   (result into y, reused)
  ln_kernel<<<TTOK, 256, 0, stream>>>(v, f_ln_g, f_ln_b, nullptr, yfbf);
  launch_gemm(stream, EPI_BIAS_GELU, yfbf, DMODEL, 0, pW1T, DMODEL, 0,
              nullptr, h1bf, 0, pb1, nullptr, 0, nullptr, 0, TTOK, PFFD, DMODEL, 1);
  launch_gemm(stream, EPI_BIAS_RES, h1bf, PFFD, 0, pW2T, PFFD, 0,
              y, nullptr, 0, pb2, v, 0, nullptr, 0, TTOK, DMODEL, PFFD, 1);

  // --- 3. gating + final FFN ---
  gate_kernel<<<4096, 256, 0, stream>>>(u, y, x, xg);
  ln_kernel<<<TTOK, 256, 0, stream>>>(xg, ln_g, ln_b, nullptr, ybf);
  launch_gemm(stream, EPI_BIAS_GELU, ybf, DMODEL, 0, Wf1T, DMODEL, 0,
              nullptr, h1bf, 0, bf1, nullptr, 0, nullptr, 0, TTOK, DFFN, DMODEL, 1);
  launch_gemm(stream, EPI_BIAS_RES, h1bf, DFFN, 0, Wf2T, DFFN, 0,
              (float*)d_out, nullptr, 0, bf2, xg, 0, nullptr, 0, TTOK, DMODEL, DFFN, 1);
}
